// MHSA_30829275250885
// MI455X (gfx1250) — matrix operand order repaired and checked by reading, not compiled, not run on hardware
//
#include <hip/hip_runtime.h>
#include <hip/hip_bf16.h>

// ---------------------------------------------------------------------------
// MHSA forward for MI455X (gfx1250, wave32, WMMA).
// All matmuls use v_wmma_f32_16x16x32_f16 (f16 inputs, f32 accumulate).
// Attention uses flash-style online softmax computing S^T so the score
// accumulator layout maps onto the B-fragment layout of the PV WMMA with a
// single cross-half shuffle (no LDS transpose needed).
// global_prefetch (gfx1250) used to run the k-loop pipeline 2 steps deep.
// ---------------------------------------------------------------------------

typedef __attribute__((ext_vector_type(16))) _Float16 v16h;
typedef __attribute__((ext_vector_type(8)))  float    v8f;

#define N_HEADS   12
#define VAL_DIM   64
#define TOKEN_DIM 768
#define SEQ       1024
#define BATCH     8
#define MTOT      (BATCH * SEQ)            // 8192 tokens
#define F3        (3 * N_HEADS * VAL_DIM)  // 2304 qkv features

static __device__ __forceinline__ v8f wmma16(v16h a, v16h b, v8f c) {
  // D = A(16x32) * B(32x16) + C, f32 accumulate
  return __builtin_amdgcn_wmma_f32_16x16x32_f16(
      /*neg_a=*/false, a, /*neg_b=*/false, b,
      /*c_mod=*/(short)0, c, /*reuse_a=*/false, /*reuse_b=*/false);
}

// A-fragment 16x32 f16: lane L holds row m = row0 + (L&15).
// element e -> k = k0 + 16*h + e          (h = L>>4)
static __device__ __forceinline__ v16h load_afrag(const _Float16* __restrict__ base,
                                                  int row0, int ld, int k0, int lane) {
  const int h = (lane >> 4) & 1;
  const _Float16* p = base + (size_t)(row0 + (lane & 15)) * ld + k0 + 16 * h;
  union { v16h v; uint4 q[2]; } u;
  u.q[0] = *reinterpret_cast<const uint4*>(p);       // k0+16h   .. +7
  u.q[1] = *reinterpret_cast<const uint4*>(p + 8);   // k0+16h+8 .. +7
  return u.v;
}

// B-fragment 32x16 f16: lane L holds column n = col0 + (L&15).
// element e -> k = k0 + 16*h + e  (contiguous 16 halves per lane)
static __device__ __forceinline__ v16h load_bfrag(const _Float16* __restrict__ base,
                                                  int col0, int ld, int k0, int lane) {
  const _Float16* p = base + (size_t)(col0 + (lane & 15)) * ld + k0 + 16 * ((lane >> 4) & 1);
  union { v16h v; uint4 q[2]; } u;
  u.q[0] = *reinterpret_cast<const uint4*>(p);
  u.q[1] = *reinterpret_cast<const uint4*>(p + 8);
  return u.v;
}

// gfx1250: lowers to global_prefetch_b8 (speculative, no counter use)
static __device__ __forceinline__ void prefetch_frag(const _Float16* __restrict__ base,
                                                     int row0, int ld, int k0, int lane) {
  const _Float16* p = base + (size_t)(row0 + (lane & 15)) * ld + k0;
  __builtin_prefetch((const void*)p, 0, 3);
}

// ---------------------------------------------------------------------------
// f32 -> f16 convert, 8 elements per thread iteration (b128 in/out)
__global__ __launch_bounds__(256) void cvt_f16_kernel(const float* __restrict__ src,
                                                      _Float16* __restrict__ dst, int n8) {
  int i = blockIdx.x * 256 + threadIdx.x;
  const int stride = gridDim.x * 256;
  for (; i < n8; i += stride) {
    const float4 a = reinterpret_cast<const float4*>(src)[2 * i];
    const float4 b = reinterpret_cast<const float4*>(src)[2 * i + 1];
    union { _Float16 h[8]; uint4 q; } u;
    u.h[0] = (_Float16)a.x; u.h[1] = (_Float16)a.y;
    u.h[2] = (_Float16)a.z; u.h[3] = (_Float16)a.w;
    u.h[4] = (_Float16)b.x; u.h[5] = (_Float16)b.y;
    u.h[6] = (_Float16)b.z; u.h[7] = (_Float16)b.w;
    reinterpret_cast<uint4*>(dst)[i] = u.q;
  }
}

// dst[n][k] = (f16) src[k][n]   (K rows x N cols -> N rows x K cols), LDS tiled
__global__ __launch_bounds__(256) void transpose_cvt_kernel(const float* __restrict__ src,
                                                            _Float16* __restrict__ dst,
                                                            int K, int N) {
  __shared__ _Float16 tile[32][33];
  const int kb = blockIdx.x * 32, nb = blockIdx.y * 32;
  const int tx = threadIdx.x & 31, ty = threadIdx.x >> 5;  // 8 row-groups
#pragma unroll
  for (int r = ty; r < 32; r += 8)
    tile[r][tx] = (_Float16)src[(size_t)(kb + r) * N + (nb + tx)];
  __syncthreads();
#pragma unroll
  for (int r = ty; r < 32; r += 8)
    dst[(size_t)(nb + r) * K + (kb + tx)] = tile[tx][r];
}

// ---------------------------------------------------------------------------
// QKV GEMM: Xh[8192][768] @ Wt[2304][768]^T(n-major) -> scatter into
//   Qh,Kh: [b][h][p][64] f16   Vt: [b][h][64][p] f16
// Block = 128 threads (4 waves); wave computes a 64x64 tile. Grid (32, 36).
__global__ __launch_bounds__(128) void qkv_gemm_kernel(const _Float16* __restrict__ Xh,
                                                       const _Float16* __restrict__ Wt,
                                                       _Float16* __restrict__ Qh,
                                                       _Float16* __restrict__ Kh,
                                                       _Float16* __restrict__ Vt) {
  const int lane = threadIdx.x & 31;
  const int wave = threadIdx.x >> 5;
  const int m0 = blockIdx.x * 256 + wave * 64;
  const int n0 = blockIdx.y * 64;

  v8f acc[4][4];
  const v8f vzero = {0.f, 0.f, 0.f, 0.f, 0.f, 0.f, 0.f, 0.f};
#pragma unroll
  for (int i = 0; i < 4; ++i)
#pragma unroll
    for (int j = 0; j < 4; ++j) acc[i][j] = vzero;

  for (int k0 = 0; k0 < TOKEN_DIM; k0 += 32) {
    if (k0 + 64 < TOKEN_DIM) {  // run the L2 pipeline 2 k-steps deep
#pragma unroll
      for (int i = 0; i < 4; ++i) prefetch_frag(Xh, m0 + 16 * i, TOKEN_DIM, k0 + 64, lane);
#pragma unroll
      for (int j = 0; j < 4; ++j) prefetch_frag(Wt, n0 + 16 * j, TOKEN_DIM, k0 + 64, lane);
    }
    v16h a[4], b[4];
#pragma unroll
    for (int i = 0; i < 4; ++i) a[i] = load_afrag(Xh, m0 + 16 * i, TOKEN_DIM, k0, lane);
#pragma unroll
    for (int j = 0; j < 4; ++j) b[j] = load_bfrag(Wt, n0 + 16 * j, TOKEN_DIM, k0, lane);
#pragma unroll
    for (int i = 0; i < 4; ++i)
#pragma unroll
      for (int j = 0; j < 4; ++j) acc[i][j] = wmma16(a[i], b[j], acc[i][j]);
  }

  const int half = lane >> 4, col = lane & 15;
#pragma unroll
  for (int i = 0; i < 4; ++i)
#pragma unroll
    for (int j = 0; j < 4; ++j)
#pragma unroll
      for (int r = 0; r < 8; ++r) {
        const int m = m0 + 16 * i + r + 8 * half;      // token index
        const int n = n0 + 16 * j + col;               // qkv feature
        const int p = m & (SEQ - 1), bb = m >> 10;
        const int head = n / (VAL_DIM * 3);
        const int rem = n - head * (VAL_DIM * 3);
        const int val = rem / 3;
        const int which = rem - val * 3;
        const _Float16 hv = (_Float16)acc[i][j][r];
        const int bh = bb * N_HEADS + head;
        if (which == 0)      Qh[((size_t)bh * SEQ + p) * VAL_DIM + val] = hv;
        else if (which == 1) Kh[((size_t)bh * SEQ + p) * VAL_DIM + val] = hv;
        else                 Vt[((size_t)bh * VAL_DIM + val) * SEQ + p] = hv;
      }
}

// ---------------------------------------------------------------------------
// Flash attention. One wave = 16 query rows of one (b,h).
// S^T = K_tile(16x32 over d) * Q^T : accumulator lane = query col, vgpr = key row.
// Online softmax stats live per-lane (per query column).
// O^T = V^T * P^T accumulated over key tiles of 32.
// Block 128 threads (4 waves); grid = 96 * 16.
__global__ __launch_bounds__(128) void attn_kernel(const _Float16* __restrict__ Qh,
                                                   const _Float16* __restrict__ Kh,
                                                   const _Float16* __restrict__ Vt,
                                                   _Float16* __restrict__ Oh) {
  const int lane = threadIdx.x & 31;
  const int wave = threadIdx.x >> 5;
  const int bh = blockIdx.x >> 4;                        // 0..95
  const int i0 = (((blockIdx.x & 15) << 2) + wave) * 16; // query row base
  const int b = bh / N_HEADS, h = bh - b * N_HEADS;

  const _Float16* Qb = Qh + (size_t)bh * SEQ * VAL_DIM;
  const _Float16* Kb = Kh + (size_t)bh * SEQ * VAL_DIM;
  const _Float16* Vb = Vt + (size_t)bh * VAL_DIM * SEQ;

  // Q as B-fragment: lane = query col i, k = d (two 32-wide chunks)
  v16h qf0 = load_bfrag(Qb, i0, VAL_DIM, 0, lane);
  v16h qf1 = load_bfrag(Qb, i0, VAL_DIM, 32, lane);

  const v8f vzero = {0.f, 0.f, 0.f, 0.f, 0.f, 0.f, 0.f, 0.f};
  v8f o[4];
#pragma unroll
  for (int a = 0; a < 4; ++a) o[a] = vzero;

  float mrun = -1e30f, lrun = 0.f;
  const float scale = 0.125f;  // 1/sqrt(64)
  const int half = lane >> 4;

  for (int j0 = 0; j0 < SEQ; j0 += 32) {
    if (j0 + 32 < SEQ) {  // prefetch next key/value tile
      prefetch_frag(Kb, j0 + 32, VAL_DIM, 0, lane);
      prefetch_frag(Kb, j0 + 48, VAL_DIM, 0, lane);
#pragma unroll
      for (int a = 0; a < 4; ++a) prefetch_frag(Vb, a * 16, SEQ, j0 + 32, lane);
    }
    // S^T tiles: s0 -> keys j0..j0+15, s1 -> keys j0+16..j0+31
    v8f s0 = vzero, s1 = vzero;
    s0 = wmma16(load_afrag(Kb, j0,      VAL_DIM, 0,  lane), qf0, s0);
    s0 = wmma16(load_afrag(Kb, j0,      VAL_DIM, 32, lane), qf1, s0);
    s1 = wmma16(load_afrag(Kb, j0 + 16, VAL_DIM, 0,  lane), qf0, s1);
    s1 = wmma16(load_afrag(Kb, j0 + 16, VAL_DIM, 32, lane), qf1, s1);

    float tmax = -1e30f;
#pragma unroll
    for (int r = 0; r < 8; ++r) {
      s0[r] *= scale; s1[r] *= scale;
      tmax = fmaxf(tmax, fmaxf(s0[r], s1[r]));
    }
    tmax = fmaxf(tmax, __shfl_xor(tmax, 16, 32));   // combine halves (same query col)
    const float mnew = fmaxf(mrun, tmax);
    const float corr = __expf(mrun - mnew);

    float p0[8], p1[8], tsum = 0.f;
#pragma unroll
    for (int r = 0; r < 8; ++r) {
      p0[r] = __expf(s0[r] - mnew);
      p1[r] = __expf(s1[r] - mnew);
      tsum += p0[r] + p1[r];
    }
    tsum += __shfl_xor(tsum, 16, 32);
    lrun = lrun * corr + tsum;
    mrun = mnew;
#pragma unroll
    for (int a = 0; a < 4; ++a)
#pragma unroll
      for (int r = 0; r < 8; ++r) o[a][r] *= corr;

    // Repack P^T into B-fragment layout (k = local key): element e -> k = 16h+e.
    // Lane (half h) holds k in {8h..8h+7} (s0) and {16+8h..23+8h} (s1);
    // the other 8-chunk lives in the partner half -> one xor-16 shuffle each.
    union { v16h v; _Float16 hh[16]; } pf;
#pragma unroll
    for (int r = 0; r < 8; ++r) {
      const float x0 = __shfl_xor(p0[r], 16, 32);
      const float x1 = __shfl_xor(p1[r], 16, 32);
      pf.hh[r]     = (_Float16)(half ? x1 : p0[r]);  // k = 16h + r
      pf.hh[8 + r] = (_Float16)(half ? p1[r] : x0);  // k = 16h + 8 + r
    }

    // O^T += V^T(16x32) * P^T(32x16) per 16-row d tile
#pragma unroll
    for (int a = 0; a < 4; ++a)
      o[a] = wmma16(load_afrag(Vb, a * 16, SEQ, j0, lane), pf.v, o[a]);
  }

  const float inv = 1.f / lrun;
  const int i = i0 + (lane & 15);
  const size_t row = ((size_t)b * SEQ + i) * (N_HEADS * VAL_DIM) + h * VAL_DIM;
#pragma unroll
  for (int a = 0; a < 4; ++a)
#pragma unroll
    for (int r = 0; r < 8; ++r) {
      const int d = a * 16 + r + 8 * half;
      Oh[row + d] = (_Float16)(o[a][r] * inv);
    }
}

// ---------------------------------------------------------------------------
// Merge GEMM: Oh[8192][768] @ Wmt[768][768]^T(n-major) -> out f32 [8192][768]
__global__ __launch_bounds__(128) void merge_gemm_kernel(const _Float16* __restrict__ Ah,
                                                         const _Float16* __restrict__ Wt,
                                                         float* __restrict__ Out) {
  const int lane = threadIdx.x & 31;
  const int wave = threadIdx.x >> 5;
  const int m0 = blockIdx.x * 256 + wave * 64;
  const int n0 = blockIdx.y * 64;

  v8f acc[4][4];
  const v8f vzero = {0.f, 0.f, 0.f, 0.f, 0.f, 0.f, 0.f, 0.f};
#pragma unroll
  for (int i = 0; i < 4; ++i)
#pragma unroll
    for (int j = 0; j < 4; ++j) acc[i][j] = vzero;

  for (int k0 = 0; k0 < TOKEN_DIM; k0 += 32) {
    if (k0 + 64 < TOKEN_DIM) {
#pragma unroll
      for (int i = 0; i < 4; ++i) prefetch_frag(Ah, m0 + 16 * i, TOKEN_DIM, k0 + 64, lane);
#pragma unroll
      for (int j = 0; j < 4; ++j) prefetch_frag(Wt, n0 + 16 * j, TOKEN_DIM, k0 + 64, lane);
    }
    v16h a[4], b[4];
#pragma unroll
    for (int i = 0; i < 4; ++i) a[i] = load_afrag(Ah, m0 + 16 * i, TOKEN_DIM, k0, lane);
#pragma unroll
    for (int j = 0; j < 4; ++j) b[j] = load_bfrag(Wt, n0 + 16 * j, TOKEN_DIM, k0, lane);
#pragma unroll
    for (int i = 0; i < 4; ++i)
#pragma unroll
      for (int j = 0; j < 4; ++j) acc[i][j] = wmma16(a[i], b[j], acc[i][j]);
  }

  const int half = lane >> 4, col = lane & 15;
#pragma unroll
  for (int i = 0; i < 4; ++i)
#pragma unroll
    for (int j = 0; j < 4; ++j)
#pragma unroll
      for (int r = 0; r < 8; ++r) {
        const int m = m0 + 16 * i + r + 8 * half;
        const int n = n0 + 16 * j + col;
        Out[(size_t)m * TOKEN_DIM + n] = acc[i][j][r];
      }
}

// ---------------------------------------------------------------------------
extern "C" void kernel_launch(void* const* d_in, const int* in_sizes, int n_in,
                              void* d_out, int out_size, void* d_ws, size_t ws_size,
                              hipStream_t stream) {
  const float* x     = (const float*)d_in[0];  // (8,1024,768)
  const float* Wqkv  = (const float*)d_in[1];  // (768,2304)
  const float* Wm    = (const float*)d_in[2];  // (768,768)
  float* out = (float*)d_out;                  // (8,1024,768) f32

  const size_t nX  = (size_t)MTOT * TOKEN_DIM;      // 6,291,456
  const size_t nQ  = (size_t)BATCH * N_HEADS * SEQ * VAL_DIM;  // 6,291,456
  const size_t nWq = (size_t)F3 * TOKEN_DIM;        // 1,769,472
  const size_t nWm = (size_t)TOKEN_DIM * TOKEN_DIM; // 589,824

  _Float16* p = (_Float16*)d_ws;
  _Float16* Xh    = p; p += nX;
  _Float16* Qh    = p; p += nQ;
  _Float16* Kh    = p; p += nQ;
  _Float16* Vth   = p; p += nQ;
  _Float16* AOh   = p; p += nX;
  _Float16* Wqkvt = p; p += nWq;
  _Float16* Wmt   = p; p += nWm;

  // 1) convert x to f16 (8 elems/thread)
  cvt_f16_kernel<<<3072, 256, 0, stream>>>(x, Xh, (int)(nX / 8));
  // 2) transpose+convert weights to N-major f16 (B-operand friendly)
  transpose_cvt_kernel<<<dim3(TOKEN_DIM / 32, F3 / 32), 256, 0, stream>>>(
      Wqkv, Wqkvt, TOKEN_DIM, F3);
  transpose_cvt_kernel<<<dim3(TOKEN_DIM / 32, TOKEN_DIM / 32), 256, 0, stream>>>(
      Wm, Wmt, TOKEN_DIM, TOKEN_DIM);
  // 3) QKV projection + head scatter (V transposed)
  qkv_gemm_kernel<<<dim3(MTOT / 256, F3 / 64), 128, 0, stream>>>(Xh, Wqkvt, Qh, Kh, Vth);
  // 4) flash attention (96 (b,h) pairs x 16 query blocks)
  attn_kernel<<<dim3(BATCH * N_HEADS * 16), 128, 0, stream>>>(Qh, Kh, Vth, AOh);
  // 5) merge projection -> f32 output
  merge_gemm_kernel<<<dim3(MTOT / 256, TOKEN_DIM / 64), 128, 0, stream>>>(AOh, Wmt, out);
}
